// ShepardsGatedAttentionBase_44555990728728
// MI455X (gfx1250) — matrix operand order repaired and checked by reading, not compiled, not run on hardware
//
#include <hip/hip_runtime.h>
#include <hip/hip_bf16.h>

#define T_SEQ  2048
#define NHEADS 16
#define DHEAD  64
#define EPSV   1e-4f

typedef __attribute__((ext_vector_type(16))) __bf16       v16bf;
typedef __attribute__((ext_vector_type(8)))  float        v8f;
typedef __attribute__((ext_vector_type(4)))  unsigned int u32x4;
typedef __attribute__((ext_vector_type(8)))  unsigned int u32x8;

union Frag  { v16bf v; u32x4 q[2]; };
union Chunk { u32x4 q; __bf16 h[8]; };

__device__ inline v8f wmma_bf16(v16bf a, v16bf b, v8f c) {
    return __builtin_amdgcn_wmma_f32_16x16x32_bf16(false, a, false, b, (short)0, c, false, false);
}

// A fragment (16x32 bf16). src row-major [rows, ld]; rows m=row0..row0+15, k=k0..k0+31.
// Lane layout (ISA 7.12.2): lane m=lane&15, half h: K in {8h..8h+7} U {16+8h..16+8h+7}.
__device__ inline v16bf load_fragA(const __bf16* base, int ld, int row0, int k0, int lane) {
    int m = row0 + (lane & 15);
    int h = (lane >> 4) & 1;
    const __bf16* p = base + (size_t)m * ld + k0 + h * 8;
    Frag f;
    f.q[0] = *(const u32x4*)(p);
    f.q[1] = *(const u32x4*)(p + 16);
    return f.v;
}

// B fragment (32x16 bf16) with B[k][n] = src[n0+n][k0+k], src row-major ld.
// Lane layout: lane n=lane&15, half h: K in {8h..8h+7} U {16+8h..16+8h+7}.
__device__ inline v16bf load_fragB(const __bf16* base, int ld, int n0, int k0, int lane) {
    int n = n0 + (lane & 15);
    int h = (lane >> 4) & 1;
    const __bf16* p = base + (size_t)n * ld + k0 + h * 8;
    Frag f;
    f.q[0] = *(const u32x4*)(p);
    f.q[1] = *(const u32x4*)(p + 16);
    return f.v;
}

__global__ void cvt_f32_bf16(const float* __restrict__ s, __bf16* __restrict__ d, int n) {
    int i = blockIdx.x * blockDim.x + threadIdx.x;
    int stride = gridDim.x * blockDim.x;
    for (; i < n; i += stride) d[i] = (__bf16)s[i];
}

// Per-row ||q||^2 and ||k||^2 over Pb [b,h,t,256]; one thread per (b,h,t) row.
__global__ __launch_bounds__(256)
void norms_kernel(const __bf16* __restrict__ Pb, float* __restrict__ Q2, float* __restrict__ K2) {
    int row = blockIdx.x * blockDim.x + threadIdx.x;
    const __bf16* p = Pb + (size_t)row * 256;
    float q = 0.f, k = 0.f;
#pragma unroll
    for (int c = 0; c < 8; ++c) {
        Chunk cq; cq.q = *(const u32x4*)(p + c * 8);
        Chunk ck; ck.q = *(const u32x4*)(p + DHEAD + c * 8);
#pragma unroll
        for (int j = 0; j < 8; ++j) {
            float vq = (float)cq.h[j]; q += vq * vq;
            float vk = (float)ck.h[j]; k += vk * vk;
        }
    }
    Q2[row] = q;
    K2[row] = k;
}

// P = X @ W_in^T + b_in, scattered into [b, h, t, 256] bf16 layout.
__global__ __launch_bounds__(256)
void gemm_in_kernel(const __bf16* __restrict__ A, const __bf16* __restrict__ Bm,
                    const float* __restrict__ bias, __bf16* __restrict__ Pb, int Kdim) {
    int lane = threadIdx.x & 31;
    int wv   = threadIdx.x >> 5;
    int m0 = blockIdx.x * 64  + (wv >> 2) * 32;
    int n0 = blockIdx.y * 256 + (wv & 3) * 64;
    v8f acc[8] = {};
    for (int k0 = 0; k0 < Kdim; k0 += 32) {
        v16bf a0 = load_fragA(A, Kdim, m0,      k0, lane);
        v16bf a1 = load_fragA(A, Kdim, m0 + 16, k0, lane);
#pragma unroll
        for (int j = 0; j < 4; ++j) {
            v16bf b = load_fragB(Bm, Kdim, n0 + j * 16, k0, lane);
            acc[j]     = wmma_bf16(a0, b, acc[j]);
            acc[4 + j] = wmma_bf16(a1, b, acc[4 + j]);
        }
    }
    int h = (lane >> 4) & 1;
#pragma unroll
    for (int j = 0; j < 4; ++j) {
        int e = n0 + j * 16 + (lane & 15);
        float bv = bias[e];
        int hh = e >> 8, c = e & 255;
#pragma unroll
        for (int mi = 0; mi < 2; ++mi) {
#pragma unroll
            for (int r = 0; r < 8; ++r) {
                int m  = m0 + mi * 16 + r + 8 * h;
                int bi = m >> 11, t = m & (T_SEQ - 1);
                Pb[(((size_t)(bi * NHEADS + hh)) * T_SEQ + t) * 256 + c] =
                    (__bf16)(acc[mi * 4 + j][r] + bv);
            }
        }
    }
}

// Y = H @ W_out^T + b_out (fp32 out).
__global__ __launch_bounds__(256)
void gemm_out_kernel(const __bf16* __restrict__ A, const __bf16* __restrict__ Bm,
                     const float* __restrict__ bias, float* __restrict__ Y,
                     int Kdim, int N) {
    int lane = threadIdx.x & 31;
    int wv   = threadIdx.x >> 5;
    int m0 = blockIdx.x * 64  + (wv >> 2) * 32;
    int n0 = blockIdx.y * 256 + (wv & 3) * 64;
    v8f acc[8] = {};
    for (int k0 = 0; k0 < Kdim; k0 += 32) {
        v16bf a0 = load_fragA(A, Kdim, m0,      k0, lane);
        v16bf a1 = load_fragA(A, Kdim, m0 + 16, k0, lane);
#pragma unroll
        for (int j = 0; j < 4; ++j) {
            v16bf b = load_fragB(Bm, Kdim, n0 + j * 16, k0, lane);
            acc[j]     = wmma_bf16(a0, b, acc[j]);
            acc[4 + j] = wmma_bf16(a1, b, acc[4 + j]);
        }
    }
    int h = (lane >> 4) & 1;
#pragma unroll
    for (int j = 0; j < 4; ++j) {
        int e = n0 + j * 16 + (lane & 15);
        float bv = bias[e];
#pragma unroll
        for (int mi = 0; mi < 2; ++mi) {
#pragma unroll
            for (int r = 0; r < 8; ++r) {
                int m = m0 + mi * 16 + r + 8 * h;
                Y[(size_t)m * N + e] = acc[mi * 4 + j][r] + bv;
            }
        }
    }
}

// Fused Shepard attention + gating. One block = one (b,h) x 128-query tile.
// K tile staged via Tensor Data Mover (async DMA), V tile transposed by VALU path.
__global__ __launch_bounds__(256)
void shepard_attn_kernel(const __bf16* __restrict__ Pb,
                         const float* __restrict__ Q2, const float* __restrict__ K2,
                         __bf16* __restrict__ Hb) {
    __shared__ __align__(16) float  k2s[T_SEQ];
    __shared__               float  q2s[128];
    __shared__ __align__(16) __bf16 Ks[32 * 64];     // K tile, row-major [k][d] (TDM dest)
    __shared__ __align__(16) __bf16 Vt[64 * 32];     // V tile, transposed [d][k]
    __shared__ __align__(16) __bf16 wsc[8][16 * 32]; // per-wave W-tile relayout scratch

    int tid  = threadIdx.x;
    int lane = tid & 31;
    int wv   = tid >> 5;
    int h    = (lane >> 4) & 1;
    int bIdx = blockIdx.z, hIdx = blockIdx.y;
    int q0blk = blockIdx.x * 128;
    const __bf16* Pbh = Pb + ((size_t)(bIdx * NHEADS + hIdx)) * T_SEQ * 256;
    const float* Q2bh = Q2 + (size_t)(bIdx * NHEADS + hIdx) * T_SEQ;
    const float* K2bh = K2 + (size_t)(bIdx * NHEADS + hIdx) * T_SEQ;

    // pull precomputed norms into LDS
    for (int i = tid; i < T_SEQ; i += 256) k2s[i] = K2bh[i];
    if (tid < 128) q2s[tid] = Q2bh[q0blk + tid];
    __syncthreads();

    int qw = q0blk + wv * 16;
    v16bf qf0 = load_fragA(Pbh, 256, qw, 0,  lane);
    v16bf qf1 = load_fragA(Pbh, 256, qw, 32, lane);

    v16bf ones;
#pragma unroll
    for (int i = 0; i < 16; ++i) ones[i] = (__bf16)1.0f;

    v8f accA[4] = {};
    v8f accR = {};
    __bf16* wrow = &wsc[wv][0];

    for (int kt = 0; kt < T_SEQ; kt += 32) {
        __syncthreads();   // prior iteration's Ks/Vt reads complete

        // ---- K tile via TDM: 2D tile 64(d) x 32(keys), 2B elems, row stride 256 ----
        if (wv == 0) {
            unsigned lds_off = (unsigned)(uintptr_t)(&Ks[0]);
            unsigned long long ga =
                (unsigned long long)(uintptr_t)(Pbh + (size_t)kt * 256 + DHEAD);
            u32x4 g0 = { 1u,                                   // count=1 (valid D#)
                         lds_off,                              // lds_addr
                         (unsigned)ga,                         // global_addr[31:0]
                         (unsigned)((ga >> 32) & 0x1FFFFFFu)   // global_addr[56:32]
                             | 0x80000000u };                  // type=2 ("image")
            u32x8 g1 = { 0x00010000u,         // workgroup_mask=0, data_size=1 (2B)
                         (64u << 16),         // tensor_dim0[15:0]=64
                         (32u << 16),         // tensor_dim0[31:16]=0, tensor_dim1[15:0]=32
                         (64u << 16),         // tensor_dim1[31:16]=0, tile_dim0=64
                         32u,                 // tile_dim1=32, tile_dim2=0
                         256u,                // tensor_dim0_stride[31:0]=256
                         0u, 0u };            // stride0 hi, stride1
            u32x4 gz = { 0u, 0u, 0u, 0u };    // groups 2/3 unused (2D)
            asm volatile("tensor_load_to_lds %0, %1, %2, %3"
                         :: "s"(g0), "s"(g1), "s"(gz), "s"(gz)
                         : "memory");
        }

        // ---- V tile transpose staging (overlaps with TDM) ----
        {
            int k  = tid >> 3;
            int db = (tid & 7) << 3;
            const __bf16* vs = Pbh + (size_t)(kt + k) * 256 + 2 * DHEAD + db;
#pragma unroll
            for (int j = 0; j < 8; ++j) Vt[(db + j) * 32 + k] = vs[j];
            // prefetch next KV tile rows
            if (kt + 32 < T_SEQ)
                __builtin_prefetch(vs + 32 * 256 - DHEAD, 0, 0);
        }

        if (wv == 0) __builtin_amdgcn_s_wait_tensorcnt(0);
        __syncthreads();

        // S = Q . K^T  (two 16x16 key tiles, K-dim 64 = 2 wmma each)
        v8f s0 = {}, s1 = {};
        s0 = wmma_bf16(qf0, load_fragB(Ks, 64, 0,  0,  lane), s0);
        s0 = wmma_bf16(qf1, load_fragB(Ks, 64, 0,  32, lane), s0);
        s1 = wmma_bf16(qf0, load_fragB(Ks, 64, 16, 0,  lane), s1);
        s1 = wmma_bf16(qf1, load_fragB(Ks, 64, 16, 32, lane), s1);

        // Shepard weights, written into A-fragment scratch
        float k2a = k2s[kt + (lane & 15)];
        float k2b = k2s[kt + 16 + (lane & 15)];
#pragma unroll
        for (int r = 0; r < 8; ++r) {
            int m = r + 8 * h;
            float q2v = q2s[wv * 16 + m];
            float d0 = fmaxf(q2v + k2a - 2.f * s0[r], 0.f);
            float d1 = fmaxf(q2v + k2b - 2.f * s1[r], 0.f);
            float t0 = EPSV + sqrtf(d0);
            float t1 = EPSV + sqrtf(d1);
            wrow[m * 32 + (lane & 15)]      = (__bf16)(1.f / (t0 * t0));
            wrow[m * 32 + 16 + (lane & 15)] = (__bf16)(1.f / (t1 * t1));
        }
        __builtin_amdgcn_wave_barrier();        // DS ops are in-order per wave
        v16bf wf = load_fragA(wrow, 32, 0, 0, lane);
        __builtin_amdgcn_wave_barrier();

        // A += W . V ; rowsum += W . ones  (same C layout as accA)
#pragma unroll
        for (int tn = 0; tn < 4; ++tn)
            accA[tn] = wmma_bf16(wf, load_fragB(Vt, 32, tn * 16, 0, lane), accA[tn]);
        accR = wmma_bf16(wf, ones, accR);
    }

    // normalize, gate with G, scatter to H [b,t, h*dh]
#pragma unroll
    for (int r = 0; r < 8; ++r) {
        int m = r + 8 * h;
        int q = qw + m;
        float inv = 1.f / (EPSV + accR[r]);
#pragma unroll
        for (int tn = 0; tn < 4; ++tn) {
            int dcol = tn * 16 + (lane & 15);
            float a = accA[tn][r] * inv;
            float g = (float)Pbh[(size_t)q * 256 + 3 * DHEAD + dcol];
            Hb[((size_t)(bIdx * T_SEQ + q)) * (NHEADS * DHEAD) + hIdx * DHEAD + dcol] =
                (__bf16)(a * g);
        }
    }
}

extern "C" void kernel_launch(void* const* d_in, const int* in_sizes, int n_in,
                              void* d_out, int out_size, void* d_ws, size_t ws_size,
                              hipStream_t stream) {
    const float* X     = (const float*)d_in[0];
    const float* W_in  = (const float*)d_in[1];
    const float* b_in  = (const float*)d_in[2];
    const float* W_out = (const float*)d_in[3];
    const float* b_out = (const float*)d_in[4];
    float* Y = (float*)d_out;

    // workspace: Xb 8MB | Wib 8MB | Wob 2MB | Pb 33.5MB | Hb 8MB | Q2 .25MB | K2 .25MB
    __bf16* Xb  = (__bf16*)d_ws;
    __bf16* Wib = Xb  + (size_t)4096 * 1024;
    __bf16* Wob = Wib + (size_t)4096 * 1024;
    __bf16* Pb  = Wob + (size_t)1024 * 1024;
    __bf16* Hb  = Pb  + (size_t)2 * NHEADS * T_SEQ * 256;
    float*  Q2  = (float*)(Hb + (size_t)2 * T_SEQ * 1024);
    float*  K2  = Q2 + (size_t)2 * NHEADS * T_SEQ;

    cvt_f32_bf16<<<2048, 256, 0, stream>>>(X,     Xb,  4096 * 1024);
    cvt_f32_bf16<<<2048, 256, 0, stream>>>(W_in,  Wib, 4096 * 1024);
    cvt_f32_bf16<<<512,  256, 0, stream>>>(W_out, Wob, 1024 * 1024);

    // P = X W_in^T + b_in : M=4096, N=4096, K=1024
    gemm_in_kernel<<<dim3(64, 16), 256, 0, stream>>>(Xb, Wib, b_in, Pb, 1024);

    // per-row Q/K norms: 2*16*2048 rows
    norms_kernel<<<(2 * NHEADS * T_SEQ) / 256, 256, 0, stream>>>(Pb, Q2, K2);

    // fused Shepard attention + gating: grid (qtiles=16, heads=16, batch=2)
    shepard_attn_kernel<<<dim3(T_SEQ / 128, NHEADS, 2), 256, 0, stream>>>(Pb, Q2, K2, Hb);

    // Y = H W_out^T + b_out : M=4096, N=1024, K=1024
    gemm_out_kernel<<<dim3(64, 4), 256, 0, stream>>>(Hb, Wob, b_out, Y, 1024, 1024);
}
